// GroupedQueryAttention_26843545600192
// MI455X (gfx1250) — matrix operand order repaired and checked by reading, not compiled, not run on hardware
//
#include <hip/hip_runtime.h>
#include <hip/hip_bf16.h>

// ---------------------------------------------------------------------------
// GQA attention layer for MI455X (gfx1250, wave32, WMMA).
// Strategy: convert all GEMM operands to f16 once (bandwidth-cheap), then every
// GEMM/attention tile staging is a pure 16-byte global->LDS copy, and all
// matrix math runs on v_wmma_f32_16x16x32_f16 with f32 accumulation.
// GEMM: 64x256 block tile, double-buffered LDS, one barrier per K-step,
// 8 WMMA/wave/K-step. Next-tile registers live only across the barrier
// (never across the WMMA block) so the kernel stays spill-free.
// ---------------------------------------------------------------------------

typedef __attribute__((ext_vector_type(16))) _Float16 v16h;
typedef __attribute__((ext_vector_type(4)))  _Float16 v4h;
typedef __attribute__((ext_vector_type(8)))  float    v8f;

namespace cfg {
constexpr int S    = 2048;
constexpr int DIN  = 2048;
constexpr int H    = 32;
constexpr int G    = 8;
constexpr int GS   = H / G;      // 4
constexpr int HD   = 128;
constexpr int DOUT = H * HD;     // 4096
constexpr int KVD  = G * HD;     // 1024
constexpr float EPS = 1e-6f;
}

// ---- WMMA fragment loaders (ISA 7.12.2 wave32 layouts) ---------------------
// A (16x32 f16): lane l -> row (l&15); K chunks {kh*8.., 16+kh*8..}, kh=l>>4.
__device__ inline v16h load_a_frag(const _Float16* base, int ld) {
  const int l  = threadIdx.x & 31;
  const int r  = l & 15;
  const int kh = l >> 4;
  v16h a;
#pragma unroll
  for (int j = 0; j < 8; ++j) {
    a[j]     = base[r * ld + kh * 8 + j];
    a[8 + j] = base[r * ld + 16 + kh * 8 + j];
  }
  return a;
}

// B (32x16 f16), row-major [K][N] in LDS: lane -> col (l&15); K chunks {kh*8.., 16+kh*8..}, kh=l>>4.
__device__ inline v16h load_b_frag(const _Float16* base, int ld) {
  const int l = threadIdx.x & 31;
  const int n = l & 15, kh = l >> 4;
  v16h b;
#pragma unroll
  for (int j = 0; j < 8; ++j) {
    b[j]     = base[(kh * 8 + j) * ld + n];
    b[8 + j] = base[(16 + kh * 8 + j) * ld + n];
  }
  return b;
}

// B fragment from a row-major [N][K] tile (logical transpose, for Q*K^T).
__device__ inline v16h load_bT_frag(const _Float16* base, int ld) {
  const int l = threadIdx.x & 31;
  const int n = l & 15, kh = l >> 4;
  v16h b;
#pragma unroll
  for (int j = 0; j < 8; ++j) {
    b[j]     = base[n * ld + kh * 8 + j];
    b[8 + j] = base[n * ld + 16 + kh * 8 + j];
  }
  return b;
}

// ---------------------------------------------------------------------------
// f32 -> f16 bulk convert, 4 elements/thread (b128 load, b64 store).
// ---------------------------------------------------------------------------
__global__ __launch_bounds__(256) void cvt_f16(
    const float* __restrict__ in, _Float16* __restrict__ out, int n4) {
  const int idx = blockIdx.x * 256 + threadIdx.x;
  if (idx >= n4) return;
  const float4 v = ((const float4*)in)[idx];
  v4h o;
  o[0] = (_Float16)v.x; o[1] = (_Float16)v.y;
  o[2] = (_Float16)v.z; o[3] = (_Float16)v.w;
  ((v4h*)out)[idx] = o;
}

// ---------------------------------------------------------------------------
// Tiled GEMM: C[M,N](f32) = A[M,K](f16) @ B[K,N](f16), row-major.
// Block: 256 threads (8 waves as 2x4), tile 64x256, K-step 32, double-buffered
// LDS. Each wave owns a 32x64 output patch = 2x4 WMMA tiles (8 WMMA/K-step).
// Steady state: one __syncthreads per K-step. Next-tile loads are issued after
// the WMMA block and retired into LDS right after the next barrier, so the
// staging registers never live across the compute (no spills). Operands are
// L2-resident (192 MB L2 >> 24 MB working set), so the post-barrier load wait
// is short and hidden by the block's 8 waves.
// Requires K % 64 == 0 and K >= 64 (true for all call sites: 2048/4096).
// ---------------------------------------------------------------------------
#define GBM 64
#define GBN 256
#define GBK 32

__global__ __launch_bounds__(256, 1) void gemm_h(
    const _Float16* __restrict__ A, const _Float16* __restrict__ B,
    float* __restrict__ C, int M, int N, int K) {
  __shared__ _Float16 As[2][GBM][GBK + 8];   // row stride 80B  (16B aligned)
  __shared__ _Float16 Bs[2][GBK][GBN + 8];   // row stride 528B (16B aligned)

  const int tid  = threadIdx.x;
  const int wave = tid >> 5;
  const int wm   = wave & 1;    // 2 waves along M (32 rows each)
  const int wn   = wave >> 1;   // 4 waves along N (64 cols each)
  const int m0   = blockIdx.y * GBM;
  const int n0   = blockIdx.x * GBN;

  // Per-thread 16B chunk assignment (compile-time shapes).
  const int ar = tid >> 2, ac = (tid & 3) * 8;     // A: 64x32  = 256 chunks (1/thr)
  const int br = tid >> 5, bc = (tid & 31) * 8;    // B: 32x256 = 1024 chunks (4/thr)

  // Pointer-carried addressing: no per-step 64-bit multiplies.
  const _Float16* aptr = A + (size_t)(m0 + ar) * K + ac;
  const _Float16* bptr = B + (size_t)br * N + n0 + bc;
  const size_t brow  = (size_t)8 * N;      // 8 K-rows apart within a tile
  const size_t bstep = (size_t)GBK * N;    // one K-step of B

  uint4 ra, rb[4];
  auto gload = [&]() {
    ra = *(const uint4*)aptr;
    aptr += GBK;
#pragma unroll
    for (int j = 0; j < 4; ++j)
      rb[j] = *(const uint4*)(bptr + (size_t)j * brow);
    bptr += bstep;
    __builtin_prefetch(aptr, 0, 1);
  };
  auto lstore = [&](int buf) {
    *(uint4*)&As[buf][ar][ac] = ra;
#pragma unroll
    for (int j = 0; j < 4; ++j)
      *(uint4*)&Bs[buf][br + j * 8][bc] = rb[j];
  };

  // Prologue: tile 0 staged in LDS buffer 0, tile 1 held in registers.
  gload();
  lstore(0);
  gload();

  v8f acc[2][4] = {};

  auto kstep = [&](int cur, int k0) {
    __syncthreads();   // buf[cur] ready; everyone done reading buf[cur^1]
    // Retire the register tile immediately: rb dies before any WMMA issues.
    if (k0 + GBK < K) lstore(cur ^ 1);    // stage tile k0+GBK
    v16h afr[2];
#pragma unroll
    for (int mi = 0; mi < 2; ++mi)
      afr[mi] = load_a_frag(&As[cur][wm * 32 + mi * 16][0], GBK + 8);
    // Consume each B fragment immediately: peak bfr liveness = 1 fragment.
#pragma unroll
    for (int ni = 0; ni < 4; ++ni) {
      v16h bfr = load_b_frag(&Bs[cur][0][wn * 64 + ni * 16], GBN + 8);
      acc[0][ni] = __builtin_amdgcn_wmma_f32_16x16x32_f16(
          false, afr[0], false, bfr, (short)0, acc[0][ni], false, false);
      acc[1][ni] = __builtin_amdgcn_wmma_f32_16x16x32_f16(
          false, afr[1], false, bfr, (short)0, acc[1][ni], false, false);
    }
    // Refill after compute: rb lives only across the upcoming barrier.
    if (k0 + 2 * GBK < K) gload();        // fetch tile k0+2*GBK
  };

  for (int k0 = 0; k0 < K; k0 += 2 * GBK) {
    kstep(0, k0);
    kstep(1, k0 + GBK);
  }

  // C/D layout: lane -> col (l&15), vgpr r -> row r + 8*(l>>4).
  const int l = tid & 31;
  const int col = l & 15;
  const int half = l >> 4;
#pragma unroll
  for (int mi = 0; mi < 2; ++mi)
#pragma unroll
    for (int ni = 0; ni < 4; ++ni)
#pragma unroll
      for (int r = 0; r < 8; ++r) {
        const int row = m0 + wm * 32 + mi * 16 + r + 8 * half;
        const int cc  = n0 + wn * 64 + ni * 16 + col;
        C[(size_t)row * N + cc] = acc[mi][ni][r];
      }
}

// ---------------------------------------------------------------------------
// RMSNorm + RoPE for Q: f32 in, f16 out with softmax scale folded in.
// Grid (S, H), 128 threads = one head row.
// ---------------------------------------------------------------------------
__global__ __launch_bounds__(128) void rmsrope_q(
    const float* __restrict__ q, const float* __restrict__ cosb,
    const float* __restrict__ sinb, const float* __restrict__ scale,
    _Float16* __restrict__ qh) {
  using namespace cfg;
  const int s = blockIdx.x, h = blockIdx.y, i = threadIdx.x;
  const float* row = q + (size_t)s * DOUT + h * HD;
  __shared__ float red[128];
  __shared__ float xn[128];

  const float xv = row[i];
  red[i] = xv * xv;
  __syncthreads();
  for (int off = 64; off > 0; off >>= 1) {
    if (i < off) red[i] += red[i + off];
    __syncthreads();
  }
  const float inv = rsqrtf(red[0] * (1.0f / HD) + EPS);
  const float nv  = xv * inv * scale[i];
  xn[i] = nv;
  __syncthreads();
  const float c  = cosb[s * HD + i];
  const float sn = sinb[s * HD + i];
  const float rot = (i < 64) ? -xn[i + 64] : xn[i - 64];
  const float sc  = 0.08838834764831845f;  // 1/sqrt(128)
  qh[(size_t)s * DOUT + h * HD + i] = (_Float16)((nv * c + rot * sn) * sc);
}

// RMSNorm + RoPE for K; writes next_k (f32, (G,S,HD)) and k_h (f16 copy).
__global__ __launch_bounds__(128) void rmsrope_k(
    const float* __restrict__ kraw, const float* __restrict__ cosb,
    const float* __restrict__ sinb, const float* __restrict__ scale,
    float* __restrict__ kout, _Float16* __restrict__ kh) {
  using namespace cfg;
  const int s = blockIdx.x, g = blockIdx.y, i = threadIdx.x;
  const float* row = kraw + (size_t)s * KVD + g * HD;
  __shared__ float red[128];
  __shared__ float xn[128];

  const float xv = row[i];
  red[i] = xv * xv;
  __syncthreads();
  for (int off = 64; off > 0; off >>= 1) {
    if (i < off) red[i] += red[i + off];
    __syncthreads();
  }
  const float inv = rsqrtf(red[0] * (1.0f / HD) + EPS);
  const float nv  = xv * inv * scale[i];
  xn[i] = nv;
  __syncthreads();
  const float c  = cosb[s * HD + i];
  const float sn = sinb[s * HD + i];
  const float rot = (i < 64) ? -xn[i + 64] : xn[i - 64];
  const float kv  = nv * c + rot * sn;
  const size_t o  = ((size_t)g * S + s) * HD + i;
  kout[o] = kv;
  kh[o]   = (_Float16)kv;
}

// V: (S, G*HD) f32 -> next_v (G,S,HD) f32 + f16 copy.
__global__ __launch_bounds__(128) void transpose_v(
    const float* __restrict__ vraw, float* __restrict__ vout,
    _Float16* __restrict__ vh) {
  using namespace cfg;
  const int s = blockIdx.x, g = blockIdx.y, i = threadIdx.x;
  const float v = vraw[(size_t)s * KVD + g * HD + i];
  const size_t o = ((size_t)g * S + s) * HD + i;
  vout[o] = v;
  vh[o]   = (_Float16)v;
}

// ---------------------------------------------------------------------------
// Flash attention, causal, GQA, all-f16 operands. Grid (S/64, H); 128 threads.
// Each of 4 waves owns a 16-query-row strip with a 16x128 f32 accumulator.
// ---------------------------------------------------------------------------
#define BQ 64
#define BKV 64

__global__ __launch_bounds__(128, 1) void attn_wmma(
    const _Float16* __restrict__ qh, const _Float16* __restrict__ kh,
    const _Float16* __restrict__ vh, _Float16* __restrict__ ctx) {
  using namespace cfg;
  __shared__ _Float16 Qs[BQ][HD + 8];
  __shared__ _Float16 Ks[BKV][HD + 8];
  __shared__ _Float16 Vs[BKV][HD + 8];
  __shared__ _Float16 Ps[4][16][BKV + 8];

  const int qb   = blockIdx.x;
  const int h    = blockIdx.y;
  const int g    = h / GS;
  const int tid  = threadIdx.x;
  const int wave = tid >> 5;
  const int l    = tid & 31;
  const int col  = l & 15;
  const int half = l >> 4;

  // Stage Q tile (64x128 f16): 1024 16B-chunks over 128 threads.
#pragma unroll
  for (int j = 0; j < 8; ++j) {
    const int c = tid + j * 128;
    const int r = c >> 4, c8 = (c & 15) * 8;
    *(uint4*)&Qs[r][c8] =
        *(const uint4*)(qh + (size_t)(qb * BQ + r) * DOUT + h * HD + c8);
  }
  __syncthreads();

  v16h aq[4];
#pragma unroll
  for (int kc = 0; kc < 4; ++kc)
    aq[kc] = load_a_frag(&Qs[wave * 16][kc * 32], HD + 8);

  v8f o[8] = {};
  float m[8], lsum[8];
#pragma unroll
  for (int r = 0; r < 8; ++r) { m[r] = -1e30f; lsum[r] = 0.0f; }

  for (int kt = 0; kt <= qb; ++kt) {
    const size_t tbase = ((size_t)g * S + kt * BKV) * HD;
    // K tile: batched loads into 32 regs, then LDS (after barrier).
    uint4 t[8];
#pragma unroll
    for (int j = 0; j < 8; ++j) {
      const int c = tid + j * 128;
      t[j] = *(const uint4*)(kh + tbase + (size_t)(c >> 4) * HD + (c & 15) * 8);
    }
    __syncthreads();  // all waves done with previous Ks/Vs
#pragma unroll
    for (int j = 0; j < 8; ++j) {
      const int c = tid + j * 128;
      *(uint4*)&Ks[c >> 4][(c & 15) * 8] = t[j];
    }
    // V tile: second 32-reg batch (loads overlap the K stores above).
#pragma unroll
    for (int j = 0; j < 8; ++j) {
      const int c = tid + j * 128;
      t[j] = *(const uint4*)(vh + tbase + (size_t)(c >> 4) * HD + (c & 15) * 8);
    }
#pragma unroll
    for (int j = 0; j < 8; ++j) {
      const int c = tid + j * 128;
      *(uint4*)&Vs[c >> 4][(c & 15) * 8] = t[j];
    }
    __syncthreads();

    // Scores: 16x64 strip = 4 col tiles x 4 K-chunks of WMMA.
    v8f sacc[4] = {};
#pragma unroll
    for (int c = 0; c < 4; ++c)
#pragma unroll
      for (int kc = 0; kc < 4; ++kc) {
        v16h b = load_bT_frag(&Ks[c * 16][kc * 32], HD + 8);
        sacc[c] = __builtin_amdgcn_wmma_f32_16x16x32_f16(
            false, aq[kc], false, b, (short)0, sacc[c], false, false);
      }

    // Causal mask on the diagonal tile.
    if (kt == qb) {
#pragma unroll
      for (int c = 0; c < 4; ++c)
#pragma unroll
        for (int r = 0; r < 8; ++r) {
          const int qi = wave * 16 + r + 8 * half;
          const int kj = c * 16 + col;
          if (kj > qi) sacc[c][r] = -1e30f;
        }
    }

    // Streaming softmax: row stats live across 16 lanes of each half.
    float alpha[8];
#pragma unroll
    for (int r = 0; r < 8; ++r) {
      float mr = sacc[0][r];
#pragma unroll
      for (int c = 1; c < 4; ++c) mr = fmaxf(mr, sacc[c][r]);
#pragma unroll
      for (int off = 1; off < 16; off <<= 1)
        mr = fmaxf(mr, __shfl_xor(mr, off, 32));
      const float mn = fmaxf(m[r], mr);
      alpha[r] = __expf(m[r] - mn);
      m[r] = mn;
    }
#pragma unroll
    for (int r = 0; r < 8; ++r) {
      float rs = 0.0f;
#pragma unroll
      for (int c = 0; c < 4; ++c) {
        const float p = __expf(sacc[c][r] - m[r]);
        sacc[c][r] = p;
        rs += p;
      }
#pragma unroll
      for (int off = 1; off < 16; off <<= 1) rs += __shfl_xor(rs, off, 32);
      lsum[r] = lsum[r] * alpha[r] + rs;
    }
#pragma unroll
    for (int cc = 0; cc < 8; ++cc)
#pragma unroll
      for (int r = 0; r < 8; ++r) o[cc][r] *= alpha[r];

    // Repack P (C-layout) -> A-layout via per-wave LDS strip.
#pragma unroll
    for (int c = 0; c < 4; ++c)
#pragma unroll
      for (int r = 0; r < 8; ++r)
        Ps[wave][r + 8 * half][c * 16 + col] = (_Float16)sacc[c][r];

    v16h ap[2];
#pragma unroll
    for (int tc = 0; tc < 2; ++tc)
      ap[tc] = load_a_frag(&Ps[wave][0][tc * 32], BKV + 8);

    // O += P @ V : 8 col tiles x 2 T-chunks of WMMA.
#pragma unroll
    for (int cc = 0; cc < 8; ++cc)
#pragma unroll
      for (int tc = 0; tc < 2; ++tc) {
        v16h b = load_b_frag(&Vs[tc * 32][cc * 16], HD + 8);
        o[cc] = __builtin_amdgcn_wmma_f32_16x16x32_f16(
            false, ap[tc], false, b, (short)0, o[cc], false, false);
      }
  }

  // Normalize and write f16 ctx in (S, H*HD) layout for the output projection.
#pragma unroll
  for (int cc = 0; cc < 8; ++cc)
#pragma unroll
    for (int r = 0; r < 8; ++r) {
      const int qrow = qb * BQ + wave * 16 + r + 8 * half;
      ctx[(size_t)qrow * DOUT + h * HD + cc * 16 + col] =
          (_Float16)(o[cc][r] / lsum[r]);
    }
}

// ---------------------------------------------------------------------------
// Host-side orchestration.
// ---------------------------------------------------------------------------
extern "C" void kernel_launch(void* const* d_in, const int* in_sizes, int n_in,
                              void* d_out, int out_size, void* d_ws, size_t ws_size,
                              hipStream_t stream) {
  using namespace cfg;
  (void)in_sizes; (void)n_in; (void)out_size; (void)ws_size;

  const float* x       = (const float*)d_in[0];
  // d_in[1] = mask (causal, reconstructed analytically — unused)
  const float* cosb    = (const float*)d_in[2];
  const float* sinb    = (const float*)d_in[3];
  const float* Wq      = (const float*)d_in[4];
  const float* Wk      = (const float*)d_in[5];
  const float* Wv      = (const float*)d_in[6];
  const float* Wo      = (const float*)d_in[7];
  const float* q_scale = (const float*)d_in[8];
  const float* k_scale = (const float*)d_in[9];

  float* out    = (float*)d_out;
  float* next_k = out + (size_t)S * DIN;          // (G, S, HD)
  float* next_v = next_k + (size_t)G * S * HD;    // (G, S, HD)

  // Workspace layout: f32 region, then f16 region (all 16B aligned).
  float* qbuf = (float*)d_ws;                      // S x DOUT   (f32)
  float* kraw = qbuf + (size_t)S * DOUT;           // S x KVD    (f32)
  float* vraw = kraw + (size_t)S * KVD;            // S x KVD    (f32)
  _Float16* hb   = (_Float16*)(vraw + (size_t)S * KVD);
  _Float16* x_h  = hb;                             // S x DIN
  _Float16* wq_h = x_h  + (size_t)S * DIN;         // DIN x DOUT
  _Float16* wk_h = wq_h + (size_t)DIN * DOUT;      // DIN x KVD
  _Float16* wv_h = wk_h + (size_t)DIN * KVD;       // DIN x KVD
  _Float16* wo_h = wv_h + (size_t)DIN * KVD;       // DOUT x DIN
  _Float16* q_h  = wo_h + (size_t)DOUT * DIN;      // S x DOUT (scaled)
  _Float16* k_h  = q_h  + (size_t)S * DOUT;        // (G,S,HD)
  _Float16* v_h  = k_h  + (size_t)G * S * HD;      // (G,S,HD)
  _Float16* c_h  = v_h  + (size_t)G * S * HD;      // S x DOUT

  auto cvt = [&](const float* src, _Float16* dst, size_t n) {
    const int n4 = (int)(n / 4);
    cvt_f16<<<dim3((n4 + 255) / 256), 256, 0, stream>>>(src, dst, n4);
  };

  // 0) One-time f32 -> f16 operand conversion.
  cvt(x,  x_h,  (size_t)S * DIN);
  cvt(Wq, wq_h, (size_t)DIN * DOUT);
  cvt(Wk, wk_h, (size_t)DIN * KVD);
  cvt(Wv, wv_h, (size_t)DIN * KVD);
  cvt(Wo, wo_h, (size_t)DOUT * DIN);

  // 1) QKV projections (WMMA GEMM, f16 operands, f32 out).
  gemm_h<<<dim3(DOUT / GBN, S / GBM), 256, 0, stream>>>(x_h, wq_h, qbuf, S, DOUT, DIN);
  gemm_h<<<dim3(KVD  / GBN, S / GBM), 256, 0, stream>>>(x_h, wk_h, kraw, S, KVD, DIN);
  gemm_h<<<dim3(KVD  / GBN, S / GBM), 256, 0, stream>>>(x_h, wv_h, vraw, S, KVD, DIN);

  // 2) RMSNorm + RoPE; K/V into output layout (G,S,HD) f32 + f16 copies.
  rmsrope_q<<<dim3(S, H), 128, 0, stream>>>(qbuf, cosb, sinb, q_scale, q_h);
  rmsrope_k<<<dim3(S, G), 128, 0, stream>>>(kraw, cosb, sinb, k_scale, next_k, k_h);
  transpose_v<<<dim3(S, G), 128, 0, stream>>>(vraw, next_v, v_h);

  // 3) Causal GQA flash attention (WMMA), f16 ctx out.
  attn_wmma<<<dim3(S / BQ, H), 128, 0, stream>>>(q_h, k_h, v_h, c_h);

  // 4) Output projection (WMMA GEMM).
  gemm_h<<<dim3(DIN / GBN, S / GBM), 256, 0, stream>>>(c_h, wo_h, out, S, DIN, DOUT);
}
